// AdaptiveEdgeSampler_21182778703895
// MI455X (gfx1250) — matrix operand order repaired and checked by reading, not compiled, not run on hardware
//
#include <hip/hip_runtime.h>

typedef __attribute__((ext_vector_type(16))) _Float16 v16h;
typedef __attribute__((ext_vector_type(8)))  float    v8f;
typedef __attribute__((address_space(3))) float lds_float;

// CDNA5 hardware transcendental tanh. TRANS-op hazard rule (ISA §7.4): result
// must not be consumed for 1 cycle -> keep a v_nop inside the asm since the
// compiler cannot know this is a TRANS op.
__device__ __forceinline__ float tanh_fast(float x) {
  float r;
  asm volatile("v_tanh_f32 %0, %1\n\tv_nop" : "=v"(r) : "v"(x));
  return r;
}

// ---------------------------------------------------------------------------
// Kernel 1: out = X @ W^T with D=64, via v_wmma_f32_16x16x32_f16.
// X: [M,64] f32 row-major, W: [64,64] f32 row-major.
// TRANSPOSE_OUT=false: out[m*64 + j]   (row-major, baskets)
// TRANSPOSE_OUT=true : out[j*ldT + m]  (transposed [64][M], items -> coalesced
//                                       loads in the score kernel)
// One wave per 16-row tile; 8 waves per 256-thread block.
// ---------------------------------------------------------------------------
template <bool TRANSPOSE_OUT>
__global__ void proj_wmma_kernel(const float* __restrict__ X,
                                 const float* __restrict__ W,
                                 float* __restrict__ out,
                                 int M, int ldT) {
  const int wave = threadIdx.x >> 5;
  const int lane = threadIdx.x & 31;
  const int tile = blockIdx.x * (blockDim.x >> 5) + wave;
  if (tile * 16 >= M) return;                 // wave-uniform: EXEC stays all-1s
  const int n0       = tile * 16;
  const int half_sel = lane >> 4;             // 0: lanes 0-15, 1: lanes 16-31
  const int row      = lane & 15;             // A row (M) / B col (N)

  // A fragments: rows n0..n0+15, K-chunks [0,32) and [32,64)
  v16h a0, a1;
#pragma unroll
  for (int h = 0; h < 16; ++h) {
    const int k = h + ((h >= 8) ? 8 : 0) + half_sel * 8;
    a0[h] = (_Float16)X[(size_t)(n0 + row) * 64 + k];
    a1[h] = (_Float16)X[(size_t)(n0 + row) * 64 + 32 + k];
  }

#pragma unroll
  for (int jt = 0; jt < 4; ++jt) {            // 4 output column tiles (D=64)
    v16h b0, b1;                              // B[k][j] = W[j][k]  (X @ W^T)
#pragma unroll
    for (int h = 0; h < 16; ++h) {
      const int k = h + ((h >= 8) ? 8 : 0) + half_sel * 8;
      b0[h] = (_Float16)W[(size_t)(jt * 16 + row) * 64 + k];
      b1[h] = (_Float16)W[(size_t)(jt * 16 + row) * 64 + 32 + k];
    }
    v8f c = {};
    c = __builtin_amdgcn_wmma_f32_16x16x32_f16(false, a0, false, b0, (short)0, c, false, false);
    c = __builtin_amdgcn_wmma_f32_16x16x32_f16(false, a1, false, b1, (short)0, c, false, false);

#pragma unroll
    for (int r = 0; r < 8; ++r) {             // D layout: m = r + 8*half_sel
      const int m = r + half_sel * 8;
      const int j = jt * 16 + row;
      if (TRANSPOSE_OUT)
        out[(size_t)j * ldT + n0 + m] = c[r];
      else
        out[(size_t)(n0 + m) * 64 + j] = c[r];
    }
  }
}

// ---------------------------------------------------------------------------
// Kernel 2: scores[b,n] = sum_d v[d] * tanh(bt[b,d] + itT[d,n])
// itT is [64][N]; a 64x256 f32 tile (64KB) is filled with CDNA5
// global_load_async_to_lds_b128 (ASYNCcnt, no VGPR staging), then reused by
// 16 baskets per block. bt/v accesses are wave-uniform -> scalar loads.
// ---------------------------------------------------------------------------
__global__ void scores_kernel(const float* __restrict__ itT,
                              const float* __restrict__ bt,
                              const float* __restrict__ v,
                              float* __restrict__ scores,
                              int N, int B) {
  __shared__ float tile[64][256];
  const int t  = threadIdx.x;                 // 256 threads
  const int n0 = blockIdx.x * 256;
  const int n  = n0 + t;
  const bool ok = (n < N);

  // ---- async tile fill: thread t copies 16 bytes of each of 16 d-rows ----
  const int d0  = t >> 6;                     // 0..3
  const int col = (t & 63) * 4;               // float column within tile row
  long gcol = n0 + col;                       // N % 4 == 0: clamped groups are
  if (gcol > (long)N - 4) gcol = (long)N - 4; // entirely in unused columns
  const unsigned lds_base = (unsigned)(uintptr_t)(lds_float*)&tile[0][0];
#pragma unroll
  for (int dc = 0; dc < 16; ++dc) {
    const int d = dc * 4 + d0;
    const unsigned long long ga =
        (unsigned long long)(uintptr_t)(itT + (size_t)d * N + gcol);
    const unsigned la = lds_base + (unsigned)((d * 256 + col) * 4);
    asm volatile("global_load_async_to_lds_b128 %0, %1, off"
                 :: "v"(la), "v"(ga) : "memory");
  }
  asm volatile("s_wait_asynccnt 0x0" ::: "memory");
  __syncthreads();

  const int b0 = blockIdx.y * 16;
  const int b1 = min(b0 + 16, B);
  for (int b = b0; b < b1; ++b) {
    const float* btr = bt + (size_t)b * 64;
    float acc = 0.f;
#pragma unroll
    for (int d = 0; d < 64; ++d)
      acc += v[d] * tanh_fast(btr[d] + tile[d][t]);
    if (ok) scores[(size_t)b * N + n] = acc;
  }
}

// ---------------------------------------------------------------------------
// Kernel 3: per-row top-k (max) and bottom-k (min) indices.
// One 1024-thread block per basket row; the whole 50000-float row is staged
// in LDS (~209KB, possible thanks to CDNA5's 320KB/WGP). k argmax passes
// (clobber winner with -inf), restore the k clobbered entries, then k argmin
// passes. Ties -> lowest index.
// ---------------------------------------------------------------------------
__global__ void topk_kernel(const float* __restrict__ scores,
                            const int* __restrict__ kptr,
                            int* __restrict__ out, int N, int B) {
  extern __shared__ float smem[];
  const int tid = threadIdx.x;
  const int bs  = blockDim.x;
  const int b   = blockIdx.x;
  const int k   = *kptr;

  float* s      = smem;                       // N floats
  float* rv     = smem + N;                   // bs floats
  int*   ri     = (int*)(rv + bs);            // bs ints
  int*   chosen = ri + bs;                    // k ints (<=256)

  const float* row = scores + (size_t)b * N;
  for (int i = tid; i < N; i += bs) s[i] = row[i];
  __syncthreads();

  // ---- top-k largest ----
  for (int it = 0; it < k; ++it) {
    float bv = -3.402823466e+38f; int bi = 0x7FFFFFFF;
    for (int i = tid; i < N; i += bs) {
      const float x = s[i];
      if (x > bv || (x == bv && i < bi)) { bv = x; bi = i; }
    }
    rv[tid] = bv; ri[tid] = bi;
    __syncthreads();
    for (int off = bs >> 1; off > 0; off >>= 1) {
      if (tid < off) {
        const float ov = rv[tid + off]; const int oi = ri[tid + off];
        if (ov > rv[tid] || (ov == rv[tid] && oi < ri[tid])) { rv[tid] = ov; ri[tid] = oi; }
      }
      __syncthreads();
    }
    if (tid == 0) {
      const int win = ri[0];
      out[(size_t)b * k + it] = win;
      chosen[it] = win;
      s[win] = -3.402823466e+38f;
    }
    __syncthreads();
  }

  // restore clobbered entries so the min pass sees original scores
  for (int i = tid; i < k; i += bs) s[chosen[i]] = row[chosen[i]];
  __syncthreads();

  // ---- top-k smallest ----
  for (int it = 0; it < k; ++it) {
    float bv = 3.402823466e+38f; int bi = 0x7FFFFFFF;
    for (int i = tid; i < N; i += bs) {
      const float x = s[i];
      if (x < bv || (x == bv && i < bi)) { bv = x; bi = i; }
    }
    rv[tid] = bv; ri[tid] = bi;
    __syncthreads();
    for (int off = bs >> 1; off > 0; off >>= 1) {
      if (tid < off) {
        const float ov = rv[tid + off]; const int oi = ri[tid + off];
        if (ov < rv[tid] || (ov == rv[tid] && oi < ri[tid])) { rv[tid] = ov; ri[tid] = oi; }
      }
      __syncthreads();
    }
    if (tid == 0) {
      const int win = ri[0];
      out[(size_t)B * k + (size_t)b * k + it] = win;
      s[win] = 3.402823466e+38f;
    }
    __syncthreads();
  }
}

// ---------------------------------------------------------------------------
extern "C" void kernel_launch(void* const* d_in, const int* in_sizes, int n_in,
                              void* d_out, int out_size, void* d_ws, size_t ws_size,
                              hipStream_t stream) {
  const float* basket = (const float*)d_in[0];   // [B,64]
  const float* item   = (const float*)d_in[1];   // [N,64]
  const float* Wb     = (const float*)d_in[2];   // [64,64]
  const float* Wi     = (const float*)d_in[3];   // [64,64]
  const float* v      = (const float*)d_in[4];   // [64]
  const int*   kptr   = (const int*)d_in[5];     // scalar k

  const int D = 64;
  const int B = in_sizes[0] / D;                 // 128
  const int N = in_sizes[1] / D;                 // 50000

  // workspace: bt [B*64] | itT [64*N] | scores [B*N]   (~38.5 MB)
  float* bt     = (float*)d_ws;
  float* itT    = bt + (size_t)B * D;
  float* scores = itT + (size_t)D * N;

  // 1) projections via WMMA
  {
    const int btiles = (B + 15) / 16;
    proj_wmma_kernel<false><<<(btiles + 7) / 8, 256, 0, stream>>>(basket, Wb, bt, B, 0);
    const int itiles = (N + 15) / 16;
    proj_wmma_kernel<true><<<(itiles + 7) / 8, 256, 0, stream>>>(item, Wi, itT, N, N);
  }

  // 2) tanh-dot interaction
  {
    dim3 grid((N + 255) / 256, (B + 15) / 16);
    scores_kernel<<<grid, 256, 0, stream>>>(itT, bt, v, scores, N, B);
  }

  // 3) per-row top-k / bottom-k (row staged in 320KB WGP LDS)
  {
    const size_t shmem = (size_t)N * sizeof(float) + 1024 * (sizeof(float) + sizeof(int))
                       + 256 * sizeof(int);
    hipFuncSetAttribute((const void*)topk_kernel,
                        hipFuncAttributeMaxDynamicSharedMemorySize, (int)shmem);
    topk_kernel<<<B, 1024, shmem, stream>>>(scores, kptr, (int*)d_out, N, B);
  }
}
